// LVAE_shGLM_16097537425993
// MI455X (gfx1250) — matrix operand order repaired and checked by reading, not compiled, not run on hardware
//
#include <hip/hip_runtime.h>
#include <stdint.h>

// ---------------------------------------------------------------------------
// Fused LVAE_shGLM encoder for MI455X (gfx1250, wave32, WMMA + async DMA).
//   windows[T,199] @ W1 -> relu -> h_mid[T,256]
//   h_mid @ W2  -> mu_mid[T,4]
//   h_mid @ Wl1 -> relu -> h_leaf[T,256]
//   h_leaf @ Wl2 -> mu_leaf[T,16]
//   out[T,20] = [mu_mid | mu_leaf]
// Weights pre-converted to f16 tile images in scratch by prep kernels, then
// staged to LDS via global_load_async_to_lds_b128 (double buffered, ASYNCcnt).
// ---------------------------------------------------------------------------

typedef __attribute__((ext_vector_type(16))) _Float16 v16h;
typedef __attribute__((ext_vector_type(8)))  _Float16 v8h;
typedef __attribute__((ext_vector_type(8)))  float    v8f;

#define TPB    128                 // 4 wave32s
#define TM     64                  // rows per workgroup (16 per wave)
#define WIN    199
#define HID    256
#define K1T    7                   // ceil(199/32) K-tiles (zero padded)
#define K2T    8                   // 256/32 K-tiles
#define WT_ST  40                  // f16 row stride of transposed weight tile
#define TILE_F16   (HID * WT_ST)   // 10240 f16 per K-tile
#define TILE_BYTES (TILE_F16 * 2)  // 20480 B
#define SM_ST  264                 // f16 row stride of w2t/wl2t/hbuf
#define SMALL_F16 (16 * SM_ST)     // 4224 f16

// LDS byte offsets
#define OFF_SIG   0                                // 320 f16   =   640 B
#define OFF_W2T   640                              // 4224 f16  =  8448 B
#define OFF_WL2T  (OFF_W2T  + SMALL_F16 * 2)       // 4224 f16  =  8448 B
#define OFF_WT0   (OFF_WL2T + SMALL_F16 * 2)       // tile buf0 = 20480 B
#define OFF_HBUF  (OFF_WT0  + 2 * TILE_BYTES)      // 4*16*264 f16 = 33792 B
#define LDS_BYTES (OFF_HBUF + 4 * 16 * SM_ST * 2)  // 92288 B

// ---- CDNA5 async global->LDS DMA (ASYNCcnt) -------------------------------
__device__ __forceinline__ void async_copy_b128(uint32_t lds_addr, const void* gsrc,
                                                int nchunks, int tid) {
  const uint64_t g = (uint64_t)(uintptr_t)gsrc;
  for (int c = tid; c < nchunks; c += TPB) {
    const uint32_t l  = lds_addr + (uint32_t)c * 16u;
    const uint64_t ga = g + (uint64_t)c * 16u;
    asm volatile("global_load_async_to_lds_b128 %0, %1, off"
                 :: "v"(l), "v"(ga) : "memory");
  }
}
__device__ __forceinline__ void wait_async0() {
  asm volatile("s_wait_asynccnt 0x0" ::: "memory");
}

// B fragment (16-bit B 32x16): lane n = l&15; lanes 0-15 K=0..15, lanes 16-31
// K=16..31; element e -> K = 16*half + e (contiguous in LDS).
__device__ __forceinline__ v16h load_bfrag(const _Float16* row_base) {
  v8h lo = *(const v8h*)(row_base);
  v8h hi = *(const v8h*)(row_base + 8);
  return __builtin_shufflevector(lo, hi, 0,1,2,3,4,5,6,7,8,9,10,11,12,13,14,15);
}

// A fragment (16-bit A 16x32): lane m = l&15; element e -> K = e + 16*half
// -> two contiguous 16B chunks, compiler emits 2x ds_load_b128.
__device__ __forceinline__ v16h afrag_gather(const _Float16* base, int half) {
  v16h a;
#pragma unroll
  for (int e = 0; e < 16; ++e) {
    const int K = e + 16 * half;
    a[e] = base[K];
  }
  return a;
}

#define WMMA_F16(A, B, C) \
  __builtin_amdgcn_wmma_f32_16x16x32_f16(false, (A), false, (B), (short)0, (C), false, false)

// ---------------------------------------------------------------------------
// Prep kernels: build f16 LDS tile images in scratch (once per launch).
// ---------------------------------------------------------------------------
__global__ void prep_wtiles(const float* __restrict__ W, _Float16* __restrict__ dst,
                            int Krows, int total) {
  const int idx = blockIdx.x * 256 + threadIdx.x;
  if (idx >= total) return;
  const int kt = idx / TILE_F16;
  const int r  = idx % TILE_F16;
  const int n  = r / WT_ST;
  const int kk = r % WT_ST;
  const int k  = kt * 32 + kk;
  const float v = (kk < 32 && k < Krows) ? W[k * HID + n] : 0.0f;
  dst[idx] = (_Float16)v;
}

__global__ void prep_small(const float* __restrict__ W2, const float* __restrict__ Wl2,
                           const float* __restrict__ V_in,
                           _Float16* __restrict__ w2t, _Float16* __restrict__ wl2t,
                           _Float16* __restrict__ vin, int T) {
  const int idx = blockIdx.x * 256 + threadIdx.x;
  if (idx < SMALL_F16) {
    const int n = idx / SM_ST, k = idx % SM_ST;
    w2t [idx] = (_Float16)((k < HID && n < 4) ? W2[k * 4 + n] : 0.0f);
    wl2t[idx] = (_Float16)((k < HID)          ? Wl2[k * 16 + n] : 0.0f);
  }
  if (idx < T) vin[idx] = (_Float16)V_in[idx];
}

// ---------------------------------------------------------------------------
// Main fused kernel
// ---------------------------------------------------------------------------
__global__ void __launch_bounds__(TPB)
lvae_encoder_wmma_kernel(const _Float16* __restrict__ vin,
                         const _Float16* __restrict__ w1t_g,   // K1T tile images
                         const _Float16* __restrict__ wl1t_g,  // K2T tile images
                         const _Float16* __restrict__ w2pair_g,// w2t | wl2t (2*4224)
                         const float* __restrict__ b1, const float* __restrict__ b2,
                         const float* __restrict__ bl1, const float* __restrict__ bl2,
                         float* __restrict__ out, int T)
{
  extern __shared__ __align__(16) char smem[];
  _Float16* sig  = (_Float16*)(smem + OFF_SIG);
  _Float16* w2t  = (_Float16*)(smem + OFF_W2T);
  _Float16* wl2t = (_Float16*)(smem + OFF_WL2T);
  _Float16* hbuf = (_Float16*)(smem + OFF_HBUF);
  char*     wt_base = smem + OFF_WT0;                       // double buffer base
  const uint32_t wt_lds0 = (uint32_t)(uintptr_t)wt_base;    // LDS byte address

  const int tid  = threadIdx.x;
  const int lane = tid & 31;
  const int wv   = tid >> 5;
  const int m16  = lane & 15;
  const int half = lane >> 4;
  const int t0   = blockIdx.x * TM;

  // ---- phase 0: async-stage small weights + first W1 tile; fill signal ----
  async_copy_b128((uint32_t)(uintptr_t)(smem + OFF_W2T), w2pair_g,
                  (2 * SMALL_F16 * 2) / 16, tid);                  // 1056 chunks
  async_copy_b128(wt_lds0, w1t_g, TILE_BYTES / 16, tid);           // W1 tile 0

  for (int i = tid; i < 320; i += TPB) {
    const int j = t0 + i - (WIN / 2);
    _Float16 v = (_Float16)0.0f;
    if ((unsigned)j < (unsigned)T) v = vin[j];
    sig[i] = v;
  }
  wait_async0();
  __syncthreads();

  const v8f vzero = {};
  v8f acc[16];

  // ---- GEMM1: h_mid = relu(windows @ W1 + b1), double-buffered tiles ------
#pragma unroll
  for (int nt = 0; nt < 16; ++nt) acc[nt] = vzero;

  int cur = 0;
  for (int kt = 0; kt < K1T; ++kt) {
    if (kt + 1 < K1T)
      async_copy_b128(wt_lds0 + (uint32_t)(cur ^ 1) * TILE_BYTES,
                      (const char*)w1t_g + (size_t)(kt + 1) * TILE_BYTES,
                      TILE_BYTES / 16, tid);
    const _Float16* wt = (const _Float16*)(wt_base + (size_t)cur * TILE_BYTES);
    const int kb = kt * 32;
    const v16h a = afrag_gather(sig + wv * 16 + m16 + kb, half);
#pragma unroll
    for (int nt = 0; nt < 16; ++nt) {
      const v16h b = load_bfrag(wt + (nt * 16 + m16) * WT_ST + half * 16);
      acc[nt] = WMMA_F16(a, b, acc[nt]);
    }
    wait_async0();
    __syncthreads();
    cur ^= 1;
  }

  // prefetch Wl1 tile 0 into the free buffer, overlapped with bias/relu
  async_copy_b128(wt_lds0 + (uint32_t)cur * TILE_BYTES, wl1t_g, TILE_BYTES / 16, tid);

  _Float16* hw = hbuf + wv * 16 * SM_ST;   // per-wave h tile [16][256]
#pragma unroll
  for (int nt = 0; nt < 16; ++nt) {
    const float bv = b1[nt * 16 + m16];
#pragma unroll
    for (int r = 0; r < 8; ++r) {
      float h = acc[nt][r] + bv;           // M = r + 8*half, N = nt*16 + m16
      h = h > 0.0f ? h : 0.0f;
      hw[(r + 8 * half) * SM_ST + nt * 16 + m16] = (_Float16)h;
    }
  }
  wait_async0();
  __syncthreads();

  // ---- GEMM2 (h_mid @ Wl1) fused with GEMM3 (h_mid @ W2) ------------------
  v8f acc3 = vzero;
#pragma unroll
  for (int nt = 0; nt < 16; ++nt) acc[nt] = vzero;

  for (int kt = 0; kt < K2T; ++kt) {
    if (kt + 1 < K2T)
      async_copy_b128(wt_lds0 + (uint32_t)(cur ^ 1) * TILE_BYTES,
                      (const char*)wl1t_g + (size_t)(kt + 1) * TILE_BYTES,
                      TILE_BYTES / 16, tid);
    const _Float16* wt = (const _Float16*)(wt_base + (size_t)cur * TILE_BYTES);
    const int kb = kt * 32;
    const v16h a = afrag_gather(hw + m16 * SM_ST + kb, half);
#pragma unroll
    for (int nt = 0; nt < 16; ++nt) {
      const v16h b = load_bfrag(wt + (nt * 16 + m16) * WT_ST + half * 16);
      acc[nt] = WMMA_F16(a, b, acc[nt]);
    }
    const v16h b3 = load_bfrag(w2t + m16 * SM_ST + kb + half * 16);
    acc3 = WMMA_F16(a, b3, acc3);
    wait_async0();
    __syncthreads();
    cur ^= 1;
  }

  // mu_mid -> out[:, 0:4]
  if (m16 < 4) {
    const float bv = b2[m16];
#pragma unroll
    for (int r = 0; r < 8; ++r) {
      const int t = t0 + wv * 16 + r + 8 * half;
      if (t < T) out[t * 20 + m16] = acc3[r] + bv;
    }
  }

  // h_leaf = relu(acc2 + bl1) overwrites the per-wave tile
#pragma unroll
  for (int nt = 0; nt < 16; ++nt) {
    const float bv = bl1[nt * 16 + m16];
#pragma unroll
    for (int r = 0; r < 8; ++r) {
      float h = acc[nt][r] + bv;
      h = h > 0.0f ? h : 0.0f;
      hw[(r + 8 * half) * SM_ST + nt * 16 + m16] = (_Float16)h;
    }
  }

  // ---- GEMM4: mu_leaf = h_leaf @ Wl2 + bl2 --------------------------------
  v8f acc4 = vzero;
#pragma unroll
  for (int kt = 0; kt < K2T; ++kt) {
    const int kb = kt * 32;
    const v16h a = afrag_gather(hw + m16 * SM_ST + kb, half);
    const v16h b = load_bfrag(wl2t + m16 * SM_ST + kb + half * 16);
    acc4 = WMMA_F16(a, b, acc4);
  }
  {
    const float bv = bl2[m16];
#pragma unroll
    for (int r = 0; r < 8; ++r) {
      const int t = t0 + wv * 16 + r + 8 * half;
      if (t < T) out[t * 20 + 4 + m16] = acc4[r] + bv;
    }
  }
}

// ---------------------------------------------------------------------------
extern "C" void kernel_launch(void* const* d_in, const int* in_sizes, int n_in,
                              void* d_out, int out_size, void* d_ws, size_t ws_size,
                              hipStream_t stream) {
  const float* V   = (const float*)d_in[0];
  const float* W1  = (const float*)d_in[1];
  const float* b1  = (const float*)d_in[2];
  const float* W2  = (const float*)d_in[3];
  const float* b2  = (const float*)d_in[4];
  const float* Wl1 = (const float*)d_in[5];
  const float* bl1 = (const float*)d_in[6];
  const float* Wl2 = (const float*)d_in[7];
  const float* bl2 = (const float*)d_in[8];
  float* out = (float*)d_out;
  const int T = in_sizes[0];

  // scratch carve-up (f16 images, all 16B aligned)
  char* ws = (char*)d_ws;
  _Float16* ws_w1  = (_Float16*)(ws);                                   // 7*10240
  _Float16* ws_wl1 = (_Float16*)(ws + (size_t)K1T * TILE_BYTES);        // 8*10240
  _Float16* ws_w2  = (_Float16*)((char*)ws_wl1 + (size_t)K2T * TILE_BYTES); // 4224
  _Float16* ws_wl2 = (_Float16*)((char*)ws_w2 + SMALL_F16 * 2);         // 4224
  _Float16* ws_vin = (_Float16*)((char*)ws_wl2 + SMALL_F16 * 2);        // T

  const int tot1 = K1T * TILE_F16;
  const int tot2 = K2T * TILE_F16;
  prep_wtiles<<<(tot1 + 255) / 256, 256, 0, stream>>>(W1, ws_w1, WIN, tot1);
  prep_wtiles<<<(tot2 + 255) / 256, 256, 0, stream>>>(Wl1, ws_wl1, HID, tot2);
  const int totS = (T > SMALL_F16 ? T : SMALL_F16);
  prep_small<<<(totS + 255) / 256, 256, 0, stream>>>(W2, Wl2, V, ws_w2, ws_wl2, ws_vin, T);

  const int grid = (T + TM - 1) / TM;
  lvae_encoder_wmma_kernel<<<grid, TPB, LDS_BYTES, stream>>>(
      ws_vin, ws_w1, ws_wl1, ws_w2, b1, b2, bl1, bl2, out, T);
}
